// TAAConvLSTMCell_47837345743179
// MI455X (gfx1250) — matrix operand order repaired and checked by reading, not compiled, not run on hardware
//
#include <hip/hip_runtime.h>

typedef __attribute__((ext_vector_type(16))) _Float16 v16h;
typedef __attribute__((ext_vector_type(8)))  float    v8f;
typedef __attribute__((ext_vector_type(2)))  float    v2f;
typedef __attribute__((ext_vector_type(4)))  unsigned int u32x4;
typedef __attribute__((ext_vector_type(4)))  int      i32x4;
typedef __attribute__((ext_vector_type(8)))  int      i32x8;

union FragAB { v16h v; uint4 u[2]; };

#if __has_builtin(__builtin_amdgcn_tensor_load_to_lds) && __has_builtin(__builtin_amdgcn_s_wait_tensorcnt)
#define HAVE_TDM 1
#else
#define HAVE_TDM 0
#endif

// ---------------------------------------------------------------------------
// Fused prep: f32->f16 weight casts, [C][HW]->[HW][C] transposes, 3x3 im2col.
// One launch, segmented by flat index.
// ---------------------------------------------------------------------------
__device__ inline _Float16 im2col_elem(const float* __restrict__ in, int C, int e) {
  int kk = e % 9;
  int c  = (e / 9) % C;
  int hw = (e / (9 * C)) % 576;
  int b  = e / (9 * C * 576);
  int y = hw / 24, x = hw - y * 24;
  int iy = y + kk / 3 - 1;
  int ix = x + (kk % 3) - 1;
  float v = 0.f;
  if (iy >= 0 && iy < 24 && ix >= 0 && ix < 24)
    v = in[((long)(b * C + c)) * 576 + iy * 24 + ix];
  return (_Float16)v;
}

__global__ __launch_bounds__(256) void prep_kernel(
    const float* __restrict__ Wx_w, const float* __restrict__ conv_w,
    const float* __restrict__ q_w,  const float* __restrict__ kv_w,
    const float* __restrict__ proj_w,
    const float* __restrict__ rep,  const float* __restrict__ history,
    const float* __restrict__ inputs,
    _Float16* __restrict__ Wx16, _Float16* __restrict__ Cw16,
    _Float16* __restrict__ Qw16, _Float16* __restrict__ KVw16,
    _Float16* __restrict__ Pw16,
    _Float16* __restrict__ rep16, _Float16* __restrict__ hist16,
    _Float16* __restrict__ xcol,  _Float16* __restrict__ rcol)
{
  int i = blockIdx.x * 256 + threadIdx.x;
  if (i < 73728) { Wx16[i] = (_Float16)Wx_w[i]; return; }
  i -= 73728;
  if (i < 73728) { Cw16[i] = (_Float16)conv_w[i]; return; }
  i -= 73728;
  if (i < 8192)  { Qw16[i] = (_Float16)q_w[i]; return; }
  i -= 8192;
  if (i < 16384) { KVw16[i] = (_Float16)kv_w[i]; return; }
  i -= 16384;
  if (i < 4096)  { Pw16[i] = (_Float16)proj_w[i]; return; }
  i -= 4096;
  if (i < 147456) {  // rep: [b][64][576] -> [b][576][64] f16
    int c = i & 63; int t = i >> 6; int hw = t % 576; int bt = t / 576;
    rep16[i] = (_Float16)rep[((long)bt * 64 + c) * 576 + hw];
    return;
  }
  i -= 147456;
  if (i < 737280) {  // history: [b*5+t][64][576] -> [bt][576][64] f16
    int c = i & 63; int t = i >> 6; int hw = t % 576; int bt = t / 576;
    hist16[i] = (_Float16)history[((long)bt * 64 + c) * 576 + hw];
    return;
  }
  i -= 737280;
  if (i < 663552) { xcol[i] = im2col_elem(inputs, 32, i); return; }
  i -= 663552;
  if (i < 1327104) { rcol[i] = im2col_elem(rep, 64, i); }
}

// ---------------------------------------------------------------------------
// f16 WMMA GEMM with LDS-staged A slab (TDM async copy when available).
//   C[bt][map(m)][n] (+=) bias[m] + sum_k A[m][k] * Bt[n][k]
// Block: 192 threads = 6 waves; block owns a 32-row A slab (all K) in LDS.
// Wave w computes the 32x32 tile at (mt*32, (ng*6+w)*32): 2x2 WMMAs.
// M % 32 == 0, N = 576, K % 32 == 0. grid = batches * (M/32) * 3.
// ---------------------------------------------------------------------------
__global__ __launch_bounds__(192) void gemm_wmma_kernel(
    const _Float16* __restrict__ A, long a_bstr, int amod,
    const _Float16* __restrict__ Bt, long b_bstr,
    float* __restrict__ C, long c_bstr, int ldc,
    const float* __restrict__ bias, int bias_bstr,
    int M, int K, int grp_div, int grp_mul, int accum)
{
  extern __shared__ _Float16 alds[];  // 32 * K halfs
  const int wave = threadIdx.x >> 5;
  const int lane = threadIdx.x & 31;
  const int mtc  = M >> 5;
  int blk = blockIdx.x;
  int ng  = blk % 3;
  int mt  = (blk / 3) % mtc;
  int bt  = blk / (3 * mtc);

  const _Float16* Aslab = A + (long)a_bstr * (bt % amod) + (long)mt * 32 * K;
  const int slab_halfs = 32 * K;

#if HAVE_TDM
  if (threadIdx.x < 32) {  // wave 0 issues one TDM DMA: 1-D copy of slab
    unsigned long long ga = (unsigned long long)(uintptr_t)Aslab;
    unsigned lds_off = (unsigned)(uintptr_t)alds;  // low 32 bits = LDS offset
    u32x4 g0;
    g0[0] = 1u;                                    // count=1, user descriptor
    g0[1] = lds_off;                               // lds_addr
    g0[2] = (unsigned)(ga & 0xFFFFFFFFu);          // global_addr[31:0]
    g0[3] = (unsigned)((ga >> 32) & 0x01FFFFFFu) | (2u << 30);  // addr[56:32], type=2
    i32x8 g1 = {};
    g1[0] = (1 << 16);                             // data_size = 2 bytes
    g1[1] = (slab_halfs & 0xFFFF) << 16;           // tensor_dim0 low16  (bits 63:48)
    g1[2] = ((slab_halfs >> 16) & 0xFFFF)          // tensor_dim0 high16 (bits 79:64)
          | (1 << 16);                             // tensor_dim1 = 1    (bits 95:80)
    g1[3] = (slab_halfs & 0xFFFF) << 16;           // tile_dim0 (bits 127:112), <= 18432
    g1[5] = slab_halfs;                            // tensor_dim0_stride low32
    i32x4 gz = {};
#if defined(__clang_major__) && (__clang_major__ >= 23)
    i32x8 gz8 = {};
    __builtin_amdgcn_tensor_load_to_lds(g0, g1, gz, gz, gz8, 0);
#else
    __builtin_amdgcn_tensor_load_to_lds(g0, g1, gz, gz, 0);
#endif
    __builtin_amdgcn_s_wait_tensorcnt(0);
  }
  __syncthreads();
#else
  for (int i = threadIdx.x; i * 8 < slab_halfs; i += 192)
    ((uint4*)alds)[i] = ((const uint4*)Aslab)[i];
  __syncthreads();
#endif

  const int l15  = lane & 15;
  const int half = lane >> 4;
  const int kb8  = half * 8;
  const int nt   = ng * 6 + wave;

  const _Float16* Bp0 = Bt + (long)b_bstr * bt + (long)(nt * 32 + l15) * K;
  const _Float16* Bp1 = Bp0 + 16L * K;
  const _Float16* Al0 = alds + (long)l15 * K;
  const _Float16* Al1 = Al0 + 16L * K;

  const int mb0 = mt * 32, mb1 = mt * 32 + 16;
  const int nb0 = nt * 32, nb1 = nt * 32 + 16;

  v8f acc00, acc01, acc10, acc11;
  #pragma unroll
  for (int v = 0; v < 8; ++v) {
    int m0 = mb0 + v + half * 8;
    int m1 = mb1 + v + half * 8;
    int mm0 = m0 + (m0 / grp_div) * grp_mul;
    int mm1 = m1 + (m1 / grp_div) * grp_mul;
    float c00 = bias[bias_bstr * (bt % amod) + m0];
    float c10 = bias[bias_bstr * (bt % amod) + m1];
    float c01 = c00, c11 = c10;
    if (accum) {
      c00 += C[c_bstr * bt + (long)mm0 * ldc + nb0 + l15];
      c01 += C[c_bstr * bt + (long)mm0 * ldc + nb1 + l15];
      c10 += C[c_bstr * bt + (long)mm1 * ldc + nb0 + l15];
      c11 += C[c_bstr * bt + (long)mm1 * ldc + nb1 + l15];
    }
    acc00[v] = c00; acc01[v] = c01; acc10[v] = c10; acc11[v] = c11;
  }

  for (int ks = 0; ks < K; ks += 32) {
    FragAB fa0, fa1, fb0, fb1;
    fa0.u[0] = *(const uint4*)(Al0 + ks + kb8);
    fa0.u[1] = *(const uint4*)(Al0 + ks + 16 + kb8);
    fa1.u[0] = *(const uint4*)(Al1 + ks + kb8);
    fa1.u[1] = *(const uint4*)(Al1 + ks + 16 + kb8);
    fb0.u[0] = *(const uint4*)(Bp0 + ks + kb8);
    fb0.u[1] = *(const uint4*)(Bp0 + ks + 16 + kb8);
    fb1.u[0] = *(const uint4*)(Bp1 + ks + kb8);
    fb1.u[1] = *(const uint4*)(Bp1 + ks + 16 + kb8);
    acc00 = __builtin_amdgcn_wmma_f32_16x16x32_f16(false, fa0.v, false, fb0.v, (short)0, acc00, false, false);
    acc01 = __builtin_amdgcn_wmma_f32_16x16x32_f16(false, fa0.v, false, fb1.v, (short)0, acc01, false, false);
    acc10 = __builtin_amdgcn_wmma_f32_16x16x32_f16(false, fa1.v, false, fb0.v, (short)0, acc10, false, false);
    acc11 = __builtin_amdgcn_wmma_f32_16x16x32_f16(false, fa1.v, false, fb1.v, (short)0, acc11, false, false);
  }

  #pragma unroll
  for (int v = 0; v < 8; ++v) {
    int m0 = mb0 + v + half * 8;
    int m1 = mb1 + v + half * 8;
    int mm0 = m0 + (m0 / grp_div) * grp_mul;
    int mm1 = m1 + (m1 / grp_div) * grp_mul;
    C[c_bstr * bt + (long)mm0 * ldc + nb0 + l15] = acc00[v];
    C[c_bstr * bt + (long)mm0 * ldc + nb1 + l15] = acc01[v];
    C[c_bstr * bt + (long)mm1 * ldc + nb0 + l15] = acc10[v];
    C[c_bstr * bt + (long)mm1 * ldc + nb1 + l15] = acc11[v];
  }
}

// ---------------------------------------------------------------------------
// Flash attention, one wave per (b, g, head, 16-query tile).
// dkh = dvh = 8, keys = T*HW = 2880 in 180 tiles of 16 (never crossing t).
// S computed transposed (rows=keys, cols=queries) via V_WMMA_F32_16X16X4_F32
// so softmax stats are per-lane; P*V done as O_T = V_T x P with the P
// B-fragment assembled from registers + one shfl_xor(16).
// ---------------------------------------------------------------------------
__global__ __launch_bounds__(32) void attn_kernel(const float* __restrict__ q_std,
                                                  const float* __restrict__ kv_std,
                                                  _Float16* __restrict__ attn16)
{
  const int lane = threadIdx.x & 31;
  int tile = blockIdx.x;
  int qt = tile % 36;
  int nh = (tile / 36) & 3;
  int g  = (tile / 144) & 3;
  int b  = tile / 576;
  const int qbase = qt * 16;
  const int nidx  = lane & 15;
  const int half  = lane >> 4;
  const float scale = 0.35355339059327376f;  // 8^-0.5

  const float* qs = q_std + ((long)b * 128 + g * 32 + nh * 8) * 576;
  v2f bq0, bq1;
  {
    int k0 = half * 2;
    bq0.x = qs[(k0 + 0) * 576 + qbase + nidx] * scale;
    bq0.y = qs[(k0 + 1) * 576 + qbase + nidx] * scale;
    bq1.x = qs[(k0 + 4) * 576 + qbase + nidx] * scale;
    bq1.y = qs[(k0 + 5) * 576 + qbase + nidx] * scale;
  }

  v8f acc = {};
  float mrun = -1e30f, lrun = 0.f;

  for (int kt = 0; kt < 180; ++kt) {
    int t   = kt / 36;
    int hwb = (kt % 36) * 16;
    const float* kbp = kv_std + (((long)(b * 5 + t) * 256) + g * 64 + nh * 8) * 576;
    const float* vbp = kbp + 32 * 576;

    v2f ak0, ak1;
    {
      int kk = half * 2;
      int hw = hwb + nidx;
      ak0.x = kbp[(kk + 0) * 576 + hw];
      ak0.y = kbp[(kk + 1) * 576 + hw];
      ak1.x = kbp[(kk + 4) * 576 + hw];
      ak1.y = kbp[(kk + 5) * 576 + hw];
    }
    v8f s = {};
    s = __builtin_amdgcn_wmma_f32_16x16x4_f32(false, ak0, false, bq0, (short)0, s, false, false);
    s = __builtin_amdgcn_wmma_f32_16x16x4_f32(false, ak1, false, bq1, (short)0, s, false, false);

    float tm = s[0];
    #pragma unroll
    for (int v = 1; v < 8; ++v) tm = fmaxf(tm, s[v]);
    tm = fmaxf(tm, __shfl_xor(tm, 16, 32));
    float mnew  = fmaxf(mrun, tm);
    float alpha = __expf(mrun - mnew);
    float p[8], ps[8], psum = 0.f;
    #pragma unroll
    for (int v = 0; v < 8; ++v) { p[v] = __expf(s[v] - mnew); psum += p[v]; }
    psum += __shfl_xor(psum, 16, 32);
    lrun = lrun * alpha + psum;
    mrun = mnew;
    #pragma unroll
    for (int v = 0; v < 8; ++v) { ps[v] = __shfl_xor(p[v], 16, 32); acc[v] *= alpha; }

    #pragma unroll
    for (int kc = 0; kc < 4; ++kc) {
      v2f av, bp;
      int kr = kc * 4 + half * 2;
      float a0 = 0.f, a1 = 0.f;
      if (nidx < 8) {
        a0 = vbp[nidx * 576 + hwb + kr + 0];
        a1 = vbp[nidx * 576 + hwb + kr + 1];
      }
      av.x = a0; av.y = a1;
      const int klo0 = 4 * kc + 0, klo1 = 4 * kc + 1;
      const int khi0 = 4 * kc + 2, khi1 = 4 * kc + 3;
      float vlo0 = (klo0 < 8) ? p[klo0] : ps[klo0 - 8];
      float vlo1 = (klo1 < 8) ? p[klo1] : ps[klo1 - 8];
      float vhi0 = (khi0 >= 8) ? p[khi0 - 8] : ps[khi0];
      float vhi1 = (khi1 >= 8) ? p[khi1 - 8] : ps[khi1];
      bp.x = half ? vhi0 : vlo0;
      bp.y = half ? vhi1 : vlo1;
      acc = __builtin_amdgcn_wmma_f32_16x16x4_f32(false, av, false, bp, (short)0, acc, false, false);
    }
  }

  if (lane < 16) {
    float inv = 1.f / lrun;
    _Float16* outp = attn16 + (((long)(b * 4 + g) * 576) + qbase + lane) * 32 + nh * 8;
    #pragma unroll
    for (int v = 0; v < 8; ++v) outp[v] = (_Float16)(acc[v] * inv);
  }
}

// ---------------------------------------------------------------------------
// Pointwise LSTM gates: pre[b][gate][64][576] -> h_t, c_t
// ---------------------------------------------------------------------------
__global__ __launch_bounds__(256) void lstm_kernel(const float* __restrict__ pre,
                                                   const float* __restrict__ cprev,
                                                   const float* __restrict__ Wci,
                                                   const float* __restrict__ Wcf,
                                                   const float* __restrict__ Wco,
                                                   float* __restrict__ out)
{
  const int PER_B = 64 * 576;
  const int TOT   = 4 * PER_B;
  int idx = blockIdx.x * 256 + threadIdx.x;
  if (idx >= TOT) return;
  int b = idx / PER_B;
  int r = idx - b * PER_B;
  long base = (long)b * 4 * PER_B + r;
  float p0 = pre[base + 0 * PER_B];
  float p1 = pre[base + 1 * PER_B];
  float p2 = pre[base + 2 * PER_B];
  float p3 = pre[base + 3 * PER_B];
  float cv  = cprev[idx];
  float i_t = 1.f / (1.f + __expf(-(p0 + cv * Wci[r])));
  float f_t = 1.f / (1.f + __expf(-(p1 + cv * Wcf[r] + 1.f)));
  float c_t = f_t * cv + i_t * tanhf(p2);
  float o_t = 1.f / (1.f + __expf(-(p3 + c_t * Wco[r])));
  out[idx]       = o_t * tanhf(c_t);
  out[TOT + idx] = c_t;
}

// ---------------------------------------------------------------------------
extern "C" void kernel_launch(void* const* d_in, const int* in_sizes, int n_in,
                              void* d_out, int out_size, void* d_ws, size_t ws_size,
                              hipStream_t stream)
{
  (void)in_sizes; (void)n_in; (void)out_size; (void)ws_size;
  const float* inputs  = (const float*)d_in[0];
  const float* rep     = (const float*)d_in[1];
  const float* c_in    = (const float*)d_in[2];
  const float* history = (const float*)d_in[3];
  const float* Wx_w    = (const float*)d_in[4];
  const float* Wx_b    = (const float*)d_in[5];
  const float* conv_w  = (const float*)d_in[6];
  const float* conv_b  = (const float*)d_in[7];
  const float* q_w     = (const float*)d_in[8];
  const float* q_b     = (const float*)d_in[9];
  const float* kv_w    = (const float*)d_in[10];
  const float* kv_b    = (const float*)d_in[11];
  const float* proj_w  = (const float*)d_in[12];
  const float* proj_b  = (const float*)d_in[13];
  const float* Wci     = (const float*)d_in[14];
  const float* Wcf     = (const float*)d_in[15];
  const float* Wco     = (const float*)d_in[16];
  float* out = (float*)d_out;

  char* ws = (char*)d_ws;
  size_t off = 0;
  auto alloc = [&](size_t bytes) -> void* {
    off = (off + 255) & ~(size_t)255;
    void* p = ws + off;
    off += bytes;
    return p;
  };

  _Float16* Wx16   = (_Float16*)alloc((size_t)256 * 288 * 2);
  _Float16* Cw16   = (_Float16*)alloc((size_t)128 * 576 * 2);
  _Float16* Qw16   = (_Float16*)alloc((size_t)128 * 64 * 2);
  _Float16* KVw16  = (_Float16*)alloc((size_t)256 * 64 * 2);
  _Float16* Pw16   = (_Float16*)alloc((size_t)128 * 32 * 2);
  _Float16* rep16  = (_Float16*)alloc((size_t)4 * 576 * 64 * 2);
  _Float16* hist16 = (_Float16*)alloc((size_t)20 * 576 * 64 * 2);
  _Float16* xcol   = (_Float16*)alloc((size_t)4 * 576 * 288 * 2);
  _Float16* rcol   = (_Float16*)alloc((size_t)4 * 576 * 576 * 2);
  float*    pre    = (float*)   alloc((size_t)4 * 4 * 64 * 576 * 4);
  float*    q_std  = (float*)   alloc((size_t)4 * 128 * 576 * 4);
  float*    kv_std = (float*)   alloc((size_t)20 * 256 * 576 * 4);
  _Float16* attn16 = (_Float16*)alloc((size_t)16 * 576 * 32 * 2);

  const int IDIV = 1 << 29;  // identity map(m)

  // One fused prep launch (3,051,520 elements -> exactly 11920 blocks)
  prep_kernel<<<11920, 256, 0, stream>>>(Wx_w, conv_w, q_w, kv_w, proj_w,
                                         rep, history, inputs,
                                         Wx16, Cw16, Qw16, KVw16, Pw16,
                                         rep16, hist16, xcol, rcol);

  // xg: M=256,K=288 -> pre (base write, bias Wx_b)
  gemm_wmma_kernel<<<4 * 8 * 3, 192, 32 * 288 * 2, stream>>>(
      Wx16, 0, 1, xcol, 576L * 288, pre, (long)4 * 64 * 576, 576,
      Wx_b, 0, 256, 288, IDIV, 0, 0);
  // co: M=128,K=576 -> pre channels 0..31 (map m -> m + (m/32)*32, accumulate)
  gemm_wmma_kernel<<<4 * 4 * 3, 192, 32 * 576 * 2, stream>>>(
      Cw16, 0, 1, rcol, 576L * 576, pre, (long)4 * 64 * 576, 576,
      conv_b, 0, 128, 576, 32, 32, 1);
  // q: M=128,K=64 -> q_std[b][128][576]
  gemm_wmma_kernel<<<4 * 4 * 3, 192, 32 * 64 * 2, stream>>>(
      Qw16, 0, 1, rep16, 576L * 64, q_std, (long)128 * 576, 576,
      q_b, 0, 128, 64, IDIV, 0, 0);
  // kv: batch = b*5+t (20), M=256,K=64 -> kv_std[b][t][256][576]
  gemm_wmma_kernel<<<20 * 8 * 3, 192, 32 * 64 * 2, stream>>>(
      KVw16, 0, 1, hist16, 576L * 64, kv_std, (long)256 * 576, 576,
      kv_b, 0, 256, 64, IDIV, 0, 0);

  // flash attention -> attn16[b*4+g][HW][32] f16
  attn_kernel<<<2304, 32, 0, stream>>>(q_std, kv_std, attn16);

  // proj: batch = b*4+g (16), M=32,K=32 -> pre channels 32..63 (accumulate)
  gemm_wmma_kernel<<<16 * 1 * 3, 192, 32 * 32 * 2, stream>>>(
      Pw16, 32 * 32, 4, attn16, 576L * 32, pre + 32 * 576, (long)64 * 576, 576,
      proj_b, 32, 32, 32, IDIV, 0, 1);

  // LSTM pointwise -> h_t || c_t
  lstm_kernel<<<(4 * 64 * 576 + 255) / 256, 256, 0, stream>>>(pre, c_in, Wci, Wcf, Wco, out);
}
